// CrossAttentionGCN_22935125361089
// MI455X (gfx1250) — matrix operand order repaired and checked by reading, not compiled, not run on hardware
//
#include <hip/hip_runtime.h>

typedef __attribute__((ext_vector_type(16))) _Float16 v16h;
typedef __attribute__((ext_vector_type(8)))  float    v8f;

// Problem constants (match reference)
constexpr int NG = 4;          // graphs / sequence length
constexpr int NN = 50000;      // nodes per graph
constexpr int NE = 800000;     // edges per graph
constexpr int NB = 32;         // pooled batch segments
constexpr int FIN = 64;        // input features
constexpr int NH = 128;        // hidden
constexpr int NHEADS = 8;
constexpr int NHD = 16;        // head dim
constexpr int NNODES = 50000;  // output classes
constexpr float OUT_SCALE = 60.0f;  // HI-LO
constexpr float OUT_BIAS  = 50.0f;  // LO

// ---------------- utility kernels ----------------

__global__ void fill_f32(float* __restrict__ p, float v, long n) {
  long i = (long)blockIdx.x * blockDim.x + threadIdx.x;
  if (i < n) p[i] = v;
}

// Pack W[K,NH] (f32, row-major) into WMMA B-fragment order (f16):
// flat t = (((kc32*(NH/16) + nt)*2 + grp)*16 + ncol)*16 + i
//   maps to W[kc32*32 + 16*grp + i][nt*16 + ncol]
// so each lane's 16 halves for one (chunk, tile) are 32 contiguous bytes.
__global__ void pack_w(const float* __restrict__ W, _Float16* __restrict__ Wp, int K) {
  int t = blockIdx.x * blockDim.x + threadIdx.x;
  if (t >= K * NH) return;
  int i    = t & 15;
  int ncol = (t >> 4) & 15;
  int grp  = (t >> 8) & 1;
  int nt   = (t >> 9) & 7;     // NH/16 = 8 tiles
  int kc32 = t >> 12;
  int k = kc32 * 32 + 16 * grp + i;
  int n = nt * 16 + ncol;
  Wp[t] = (_Float16)W[(size_t)k * NH + n];
}

// deg[g][n] starts at 1.0 (self loop); add 1 per incoming edge at col
__global__ void deg_count(const int* __restrict__ ei, float* __restrict__ deg) {
  long t = (long)blockIdx.x * blockDim.x + threadIdx.x;
  if (t >= (long)NG * NE) return;
  int g = (int)(t / NE);
  int e = (int)(t % NE);
  int col = ei[(size_t)g * 2 * NE + NE + e];
  atomicAdd(&deg[(size_t)g * NN + col], 1.0f);
}

// ---------------- WMMA GEMM: C[M,NH] = A[M,K](f32) @ W(packed f16) ----------------
// block = 256 threads (8 waves); each block computes one 16-row x 128-col stripe.
// A tile (16 x K f32) is staged in LDS with async global->LDS copies; all 8 waves
// share it, wave w computes output tile columns [16w, 16w+16).
template <int K>
__global__ void wmma_gemm(const float* __restrict__ A, const _Float16* __restrict__ Wp,
                          float* __restrict__ C, long strideA, long strideC) {
  __shared__ float As[16 * K];  // K=128 -> 8KB

  const int tid  = threadIdx.x;
  const int m0   = blockIdx.x * 16;
  const float* Ag = A + (long)blockIdx.z * strideA;
  float*       Cg = C + (long)blockIdx.z * strideC;

  // ---- stage A tile into LDS via async global->LDS (B128 = 4 floats/lane) ----
  constexpr int NQ = 16 * K / 4;  // float4 transfers (256 or 512)
#pragma unroll
  for (int q = tid; q < NQ; q += 256) {
    int r  = q / (K / 4);
    int kq = (q % (K / 4)) * 4;
    unsigned long long gaddr = (unsigned long long)(const void*)(Ag + (long)(m0 + r) * K + kq);
    unsigned ldsoff = (unsigned)(uintptr_t)(&As[r * K + kq]);
    asm volatile("global_load_async_to_lds_b128 %0, %1, off"
                 :: "v"(ldsoff), "v"(gaddr) : "memory");
  }
  asm volatile("s_wait_asynccnt 0x0" ::: "memory");
  __syncthreads();

  const int lane = tid & 31;
  const int wave = tid >> 5;       // 0..7 -> output column tile
  const int n0   = wave * 16;
  const int mrow = lane & 15;
  const int grp  = lane >> 4;      // 0 or 1
  const int ncol = lane & 15;

  v8f acc = {};
#pragma unroll
  for (int kc = 0; kc < K; kc += 32) {
    // A fragment from LDS: halves 0..15 -> K=kc+16*grp+i
    const float* arow = &As[mrow * K + kc + 16 * grp];
    v16h a;
#pragma unroll
    for (int i = 0; i < 8; ++i) {
      a[i]     = (_Float16)arow[i];
      a[8 + i] = (_Float16)arow[8 + i];
    }
    // B fragment: 16 contiguous halves from packed weights
    const v16h b = *(const v16h*)(Wp +
        ((((size_t)(kc >> 5) * 8 + wave) * 2 + grp) * 16 + ncol) * 16);

    acc = __builtin_amdgcn_wmma_f32_16x16x32_f16(
        /*neg_a=*/false, a, /*neg_b=*/false, b,
        /*c_mod=*/(short)0, acc, /*reuse_a=*/false, /*reuse_b=*/false);
  }
  // D: lane col = n0+ncol, VGPR r -> row m0 + r + 8*grp
#pragma unroll
  for (int r = 0; r < 8; ++r)
    Cg[(long)(m0 + r + 8 * grp) * NH + n0 + ncol] = acc[r];
}

// ---------------- GCN edge aggregation (scatter-add) ----------------
__global__ void gcn_aggregate(const float* __restrict__ hlin, float* __restrict__ agg,
                              const int* __restrict__ ei, const float* __restrict__ deg) {
  long t = (long)blockIdx.x * blockDim.x + threadIdx.x;
  const long total = (long)NG * NE * (NH / 4);
  if (t >= total) return;
  int  c  = (int)(t & 31);
  long te = t >> 5;
  int  e  = (int)(te % NE);
  int  g  = (int)(te / NE);
  const int* eig = ei + (size_t)g * 2 * NE;
  int row = eig[e];
  int col = eig[NE + e];
  float norm = rsqrtf(deg[(size_t)g * NN + row]) * rsqrtf(deg[(size_t)g * NN + col]);
  const float4 hv = *(const float4*)(hlin + ((size_t)g * NN + row) * NH + c * 4);
  float* dst = agg + ((size_t)g * NN + col) * NH + c * 4;
  atomicAdd(dst + 0, hv.x * norm);
  atomicAdd(dst + 1, hv.y * norm);
  atomicAdd(dst + 2, hv.z * norm);
  atomicAdd(dst + 3, hv.w * norm);
}

// agg = relu(agg + hlin/deg + bias)  (in place over agg)
__global__ void gcn_combine(float* __restrict__ agg, const float* __restrict__ hlin,
                            const float* __restrict__ deg, const float* __restrict__ bias) {
  long t = (long)blockIdx.x * blockDim.x + threadIdx.x;
  const long total = (long)NG * NN * NH;
  if (t >= total) return;
  int  f  = (int)(t % NH);
  long gn = t / NH;
  float v = agg[t] + hlin[t] / deg[gn] + bias[f];
  agg[t] = v > 0.0f ? v : 0.0f;
}

// ---------------- segment mean pooling ----------------
__global__ void pool_scatter(const float* __restrict__ h, const int* __restrict__ batch,
                             float* __restrict__ s, float* __restrict__ cnt) {
  long t = (long)blockIdx.x * blockDim.x + threadIdx.x;
  const long total = (long)NG * NN * (NH / 4);
  if (t >= total) return;
  int  c  = (int)(t & 31);
  long gn = t >> 5;
  int  n  = (int)(gn % NN);
  int  g  = (int)(gn / NN);
  int  bg = batch[(size_t)g * NN + n];
  const float4 hv = *(const float4*)(h + gn * NH + c * 4);
  float* dst = s + ((size_t)g * NB + bg) * NH + c * 4;
  atomicAdd(dst + 0, hv.x);
  atomicAdd(dst + 1, hv.y);
  atomicAdd(dst + 2, hv.z);
  atomicAdd(dst + 3, hv.w);
  if (c == 0) atomicAdd(&cnt[g * NB + bg], 1.0f);
}

__global__ void pool_finalize(const float* __restrict__ s, const float* __restrict__ cnt,
                              float* __restrict__ emb) {
  int t = blockIdx.x * blockDim.x + threadIdx.x;
  if (t >= NG * NB * NH) return;
  int gb = t / NH;
  float c = cnt[gb];
  emb[t] = (c > 0.0f) ? s[t] / fmaxf(c, 1.0f) : 0.0f;
}

// ---------------- attention (tiny) ----------------
__global__ void qkv_proj(const float* __restrict__ emb, const float* __restrict__ w,
                         const float* __restrict__ b, float* __restrict__ qkv) {
  int t = blockIdx.x * blockDim.x + threadIdx.x;
  if (t >= NG * NB * 3 * NH) return;
  int i  = t % (3 * NH);
  int gb = t / (3 * NH);
  float acc = b[i];
  const float* er = emb + (size_t)gb * NH;
  const float* wr = w + (size_t)i * NH;
  for (int j = 0; j < NH; ++j) acc += er[j] * wr[j];
  qkv[(size_t)gb * 3 * NH + i] = acc;
}

__global__ void mha_attend(const float* __restrict__ qkv, float* __restrict__ ctx) {
  int t = blockIdx.x * blockDim.x + threadIdx.x;
  if (t >= NG * NB * NHEADS) return;
  int h = t % NHEADS;
  int b = (t / NHEADS) % NB;
  int g = t / (NHEADS * NB);
  const int S = 3 * NH;
  const float* q = qkv + (size_t)(g * NB + b) * S + h * NHD;
  float sc[NG];
  float mx = -1e30f;
  for (int k = 0; k < NG; ++k) {
    const float* kk = qkv + (size_t)(k * NB + b) * S + NH + h * NHD;
    float d = 0.0f;
    for (int j = 0; j < NHD; ++j) d += q[j] * kk[j];
    sc[k] = d * 0.25f;  // 1/sqrt(16)
    mx = fmaxf(mx, sc[k]);
  }
  float den = 0.0f;
  for (int k = 0; k < NG; ++k) { sc[k] = __expf(sc[k] - mx); den += sc[k]; }
  float inv = 1.0f / den;
  for (int j = 0; j < NHD; ++j) {
    float acc = 0.0f;
    for (int k = 0; k < NG; ++k)
      acc += sc[k] * inv * qkv[(size_t)(k * NB + b) * S + 2 * NH + h * NHD + j];
    ctx[(size_t)(g * NB + b) * NH + h * NHD + j] = acc;
  }
}

__global__ void outproj_pool(const float* __restrict__ ctx, const float* __restrict__ w,
                             const float* __restrict__ b, float* __restrict__ pooled) {
  int t = blockIdx.x * blockDim.x + threadIdx.x;
  if (t >= NG * NB * NH) return;
  int i  = t % NH;
  int gb = t / NH;
  int g  = gb / NB;
  float acc = b[i];
  const float* cr = ctx + (size_t)gb * NH;
  const float* wr = w + (size_t)i * NH;
  for (int j = 0; j < NH; ++j) acc += cr[j] * wr[j];
  atomicAdd(&pooled[g * NH + i], acc * (1.0f / NB));
}

// out[g][m] = (pooled[g] . lin_w[m] + lin_b[m]) * 60 + 50
__global__ void final_linear(const float* __restrict__ pooled, const float* __restrict__ lin_w,
                             const float* __restrict__ lin_b, float* __restrict__ out) {
  __shared__ float ps[NG * NH];
  for (int i = threadIdx.x; i < NG * NH; i += blockDim.x) ps[i] = pooled[i];
  __syncthreads();
  int m = blockIdx.x * blockDim.x + threadIdx.x;
  if (m >= NNODES) return;
  const float* wr = lin_w + (size_t)m * NH;
  __builtin_prefetch(wr + NH, 0, 3);
  float acc[NG] = {0.f, 0.f, 0.f, 0.f};
  for (int j = 0; j < NH; ++j) {
    float w = wr[j];
#pragma unroll
    for (int g = 0; g < NG; ++g) acc[g] += ps[g * NH + j] * w;
  }
  float bv = lin_b[m];
#pragma unroll
  for (int g = 0; g < NG; ++g)
    out[(size_t)g * NNODES + m] = (acc[g] + bv) * OUT_SCALE + OUT_BIAS;
}

// ---------------- launcher ----------------
extern "C" void kernel_launch(void* const* d_in, const int* in_sizes, int n_in,
                              void* d_out, int out_size, void* d_ws, size_t ws_size,
                              hipStream_t stream) {
  (void)in_sizes; (void)n_in; (void)out_size; (void)ws_size;
  const float* x          = (const float*)d_in[0];
  const int*   edge_index = (const int*)  d_in[1];
  const int*   batch      = (const int*)  d_in[2];
  const float* W1         = (const float*)d_in[3];
  const float* b1         = (const float*)d_in[4];
  const float* W2         = (const float*)d_in[5];
  const float* b2         = (const float*)d_in[6];
  const float* in_proj_w  = (const float*)d_in[7];
  const float* in_proj_b  = (const float*)d_in[8];
  const float* out_proj_w = (const float*)d_in[9];
  const float* out_proj_b = (const float*)d_in[10];
  const float* lin_w      = (const float*)d_in[11];
  const float* lin_b      = (const float*)d_in[12];
  float* out = (float*)d_out;

  // workspace carve
  float* buf0   = (float*)d_ws;                      // [NG*NN*NH] linear (GEMM) output
  float* buf1   = buf0   + (size_t)NG * NN * NH;     // [NG*NN*NH] aggregation / hidden
  float* deg    = buf1   + (size_t)NG * NN * NH;     // [NG*NN]
  float* spool  = deg    + (size_t)NG * NN;          // [NG*NB*NH]
  float* cnt    = spool  + (size_t)NG * NB * NH;     // [NG*NB]
  float* pooled = cnt    + (size_t)NG * NB;          // [NG*NH]
  float* emb    = pooled + (size_t)NG * NH;          // [NG*NB*NH]
  float* qkv    = emb    + (size_t)NG * NB * NH;     // [NG*NB*3NH]
  float* ctx    = qkv    + (size_t)NG * NB * 3 * NH; // [NG*NB*NH]
  _Float16* W1p = (_Float16*)(ctx + (size_t)NG * NB * NH);
  _Float16* W2p = W1p + FIN * NH;

  auto blocks = [](long total, int bs) { return dim3((unsigned)((total + bs - 1) / bs)); };

  // weights -> packed f16 B-fragment order
  pack_w<<<blocks(FIN * NH, 256), 256, 0, stream>>>(W1, W1p, FIN);
  pack_w<<<blocks(NH * NH, 256), 256, 0, stream>>>(W2, W2p, NH);

  // degrees (self-loop = 1.0 initial)
  fill_f32<<<blocks((long)NG * NN, 256), 256, 0, stream>>>(deg, 1.0f, (long)NG * NN);
  deg_count<<<blocks((long)NG * NE, 256), 256, 0, stream>>>(edge_index, deg);

  // ---- GCN layer 1 ----
  wmma_gemm<FIN><<<dim3(NN / 16, 1, NG), 256, 0, stream>>>(
      x, W1p, buf0, (long)NN * FIN, (long)NN * NH);
  fill_f32<<<blocks((long)NG * NN * NH, 256), 256, 0, stream>>>(buf1, 0.0f, (long)NG * NN * NH);
  gcn_aggregate<<<blocks((long)NG * NE * (NH / 4), 256), 256, 0, stream>>>(
      buf0, buf1, edge_index, deg);
  gcn_combine<<<blocks((long)NG * NN * NH, 256), 256, 0, stream>>>(buf1, buf0, deg, b1);

  // ---- GCN layer 2 ----
  wmma_gemm<NH><<<dim3(NN / 16, 1, NG), 256, 0, stream>>>(
      buf1, W2p, buf0, (long)NN * NH, (long)NN * NH);
  fill_f32<<<blocks((long)NG * NN * NH, 256), 256, 0, stream>>>(buf1, 0.0f, (long)NG * NN * NH);
  gcn_aggregate<<<blocks((long)NG * NE * (NH / 4), 256), 256, 0, stream>>>(
      buf0, buf1, edge_index, deg);
  gcn_combine<<<blocks((long)NG * NN * NH, 256), 256, 0, stream>>>(buf1, buf0, deg, b2);

  // ---- segment mean pool -> emb ----
  fill_f32<<<blocks((long)(NG * NB * NH + NG * NB + NG * NH), 256), 256, 0, stream>>>(
      spool, 0.0f, (long)(NG * NB * NH + NG * NB + NG * NH));
  pool_scatter<<<blocks((long)NG * NN * (NH / 4), 256), 256, 0, stream>>>(buf1, batch, spool, cnt);
  pool_finalize<<<blocks(NG * NB * NH, 256), 256, 0, stream>>>(spool, cnt, emb);

  // ---- attention ----
  qkv_proj<<<blocks(NG * NB * 3 * NH, 256), 256, 0, stream>>>(emb, in_proj_w, in_proj_b, qkv);
  mha_attend<<<blocks(NG * NB * NHEADS, 256), 256, 0, stream>>>(qkv, ctx);
  outproj_pool<<<blocks(NG * NB * NH, 256), 256, 0, stream>>>(ctx, out_proj_w, out_proj_b, pooled);

  // ---- final classifier ----
  final_linear<<<blocks(NNODES, 256), 256, 0, stream>>>(pooled, lin_w, lin_b, out);
}
